// RoutedFFN_58806692217008
// MI455X (gfx1250) — matrix operand order repaired and checked by reading, not compiled, not run on hardware
//
#include <hip/hip_runtime.h>

// ---------------------------------------------------------------------------
// RoutedFFN for MI455X (gfx1250): bf16 WMMA, MoE token grouping, 32-token
// tiles (B-fragment register reuse x2), L2-resident bf16 weights,
// fragment-layout LDS staging, atomic f32 accumulation into y.
// ---------------------------------------------------------------------------

#define IN_F   2048
#define OUT_F  8192
#define NB     16
#define BS     512           // hidden units per expert block
#define TOPK   4
#define NTOK   8192          // 2*4096 tokens
#define MTILE  32            // tokens per workgroup tile (2 x 16-row subtiles)
#define MSUB   2
#define MAXT   NTOK          // per-block list capacity
#define MAX_TILES (NTOK / MTILE)   // 256

#define XS_BYTES   (MSUB * 64 * 1024)   // 128 KB: [msub][kstep64][piece2][lane32][16B]
#define HS_BYTES   (MSUB * 16 * 1024)   //  32 KB: [msub][kstep16][piece2][lane32][16B]
#define SMEM_BYTES (XS_BYTES + HS_BYTES + MTILE * 4)

typedef __attribute__((ext_vector_type(16))) __bf16     bf16x16;
typedef __attribute__((ext_vector_type(8)))  float      f32x8;
typedef __attribute__((ext_vector_type(4)))  float      f32x4;
typedef __attribute__((ext_vector_type(4)))  unsigned   u32x4;

union AFrag { bf16x16 v; u32x4 q[2]; };

__device__ __forceinline__ unsigned short f2bf(float f) {
    unsigned u = __float_as_uint(f);
    unsigned r = (u + 0x7FFFu + ((u >> 16) & 1u)) >> 16;   // round-nearest-even
    return (unsigned short)r;
}

__device__ __forceinline__ u32x4 pack8(f32x4 a, f32x4 b) {
    u32x4 o;
    o.x = (unsigned)f2bf(a.x) | ((unsigned)f2bf(a.y) << 16);
    o.y = (unsigned)f2bf(a.z) | ((unsigned)f2bf(a.w) << 16);
    o.z = (unsigned)f2bf(b.x) | ((unsigned)f2bf(b.y) << 16);
    o.w = (unsigned)f2bf(b.z) | ((unsigned)f2bf(b.w) << 16);
    return o;
}

// -------------------- 1) fp32 -> bf16 weight conversion --------------------
__global__ void convert_bf16(const float* __restrict__ src,
                             unsigned short* __restrict__ dst, int n) {
    int i = blockIdx.x * blockDim.x + threadIdx.x;
    long base = (long)i * 8;
    if (base >= n) return;
    const f32x4* s = (const f32x4*)(src + base);
    ((u32x4*)dst)[i] = pack8(s[0], s[1]);
}

// -------------------- 2) router: top-4 expert grouping ---------------------
// One wave per token. Lanes 0-15 own one logit each; lane halves split K.
__global__ __launch_bounds__(256)
void router_topk(const float* __restrict__ x, const float* __restrict__ Wr,
                 const float* __restrict__ br,
                 int* __restrict__ cnt, int* __restrict__ list) {
    const int lane  = threadIdx.x & 31;
    const int wv    = threadIdx.x >> 5;
    const int token = blockIdx.x * 8 + wv;
    const int bsel  = lane & 15;
    const int half  = lane >> 4;

    const f32x4* xr = (const f32x4*)(x  + (size_t)token * IN_F + half * 1024);
    const f32x4* wr = (const f32x4*)(Wr + (size_t)bsel  * IN_F + half * 1024);
    float s = 0.0f;
    for (int k = 0; k < 256; ++k) {
        f32x4 a = xr[k], w = wr[k];
        s += a.x * w.x + a.y * w.y + a.z * w.z + a.w * w.w;
    }
    s += __shfl_down(s, 16, 32);                    // combine K halves
    float logit = (lane < 16) ? (s + br[bsel]) : -3.4e38f;

    // top-k of logits == top-k of softmax(logits)
    for (int t = 0; t < TOPK; ++t) {
        float m = logit;
        m = fmaxf(m, __shfl_xor(m, 16, 32));
        m = fmaxf(m, __shfl_xor(m,  8, 32));
        m = fmaxf(m, __shfl_xor(m,  4, 32));
        m = fmaxf(m, __shfl_xor(m,  2, 32));
        m = fmaxf(m, __shfl_xor(m,  1, 32));
        unsigned long long bal = __ballot(logit == m);
        int win = (int)__ffsll((long long)bal) - 1;
        if (lane == win) {
            int pos = atomicAdd(&cnt[bsel], 1);
            list[bsel * MAXT + pos] = token;
            logit = -3.4e38f;
        }
    }
}

// -------------------- 3) grouped expert FFN (bf16 WMMA) --------------------
// grid = (MAX_TILES, NB); 256 threads = 8 waves; 160KB dynamic LDS.
// Each workgroup: 32 tokens of one expert. Each B fragment feeds 2 WMMAs
// (the two 16-row M-subtiles), doubling MACs per L2 byte vs a 16-token tile.
__global__ __launch_bounds__(256)
void ffn_kernel(const float* __restrict__ x,
                const unsigned short* __restrict__ W1bf,
                const unsigned short* __restrict__ W2bf,
                const int* __restrict__ cnt, const int* __restrict__ list,
                float* __restrict__ y) {
    extern __shared__ char smem[];
    char* Xs  = smem;                         // 128 KB
    char* Hs  = smem + XS_BYTES;              //  32 KB
    int*  tok = (int*)(smem + XS_BYTES + HS_BYTES);

    const int b      = blockIdx.y;
    const int tstart = blockIdx.x * MTILE;
    const int c      = cnt[b];
    if (tstart >= c) return;                  // data-dependent tile skip
    const int nvalid = min(MTILE, c - tstart);

    const int tid = threadIdx.x;
    if (tid < MTILE) {
        int idx = tstart + tid;
        tok[tid] = list[b * MAXT + (tid < nvalid ? idx : tstart)];
    }
    __syncthreads();

    // gather 32 fp32 token rows, convert to bf16, store in A-fragment layout
    for (int p = tid; p < MTILE * 256; p += 256) {
        int row = p >> 8, k8 = p & 255;           // 8 elems (16B bf16) per piece
        const f32x4* s = (const f32x4*)(x + (size_t)tok[row] * IN_F + k8 * 8);
        u32x4 o = pack8(s[0], s[1]);
        int msub = row >> 4, r16 = row & 15;
        int kstep = k8 >> 2, id2 = k8 & 3, hi = id2 & 1, pc = id2 >> 1;
        *(u32x4*)(Xs + msub * 65536 + kstep * 1024 + pc * 512 +
                  (r16 + 16 * hi) * 16) = o;
    }
    __syncthreads();

    const int lane = tid & 31, wv = tid >> 5;
    const int lh = lane & 15, lq = lane >> 4;

    // ---- FC1: H[32,512] = X[32,2048] * W1_b^T; per wave 4 N-tiles x 2 M ----
    f32x8 acc[MSUB][4];
#pragma unroll
    for (int ms = 0; ms < MSUB; ++ms)
#pragma unroll
        for (int i = 0; i < 4; ++i) acc[ms][i] = (f32x8)0.0f;

    for (int ks = 0; ks < 64; ++ks) {
        AFrag A[MSUB];
#pragma unroll
        for (int ms = 0; ms < MSUB; ++ms) {
            A[ms].q[0] = *(const u32x4*)(Xs + ms * 65536 + ks * 1024 +       lane * 16);
            A[ms].q[1] = *(const u32x4*)(Xs + ms * 65536 + ks * 1024 + 512 + lane * 16);
        }
#pragma unroll
        for (int i = 0; i < 4; ++i) {
            int grow = b * BS + (wv * 4 + i) * 16 + lh;      // W1 row = B column
            const u32x4* bp =
                (const u32x4*)(W1bf + (size_t)grow * IN_F + ks * 32 + lq * 8);
            AFrag B; B.q[0] = bp[0]; B.q[1] = bp[2];
#pragma unroll
            for (int ms = 0; ms < MSUB; ++ms)                // B reused x2
                acc[ms][i] = __builtin_amdgcn_wmma_f32_16x16x32_bf16(
                    false, A[ms].v, false, B.v, (short)0, acc[ms][i], false, false);
        }
    }

    // ---- exact GELU + scatter into Hs (A-fragment order, K = j) ----
#pragma unroll
    for (int ms = 0; ms < MSUB; ++ms) {
#pragma unroll
        for (int i = 0; i < 4; ++i) {
            int jn = (wv * 4 + i) * 16 + lh;             // this lane's column j
            int kstep = jn >> 5, klo = jn & 31;
            int id2 = klo >> 3, hi = id2 & 1, pc = id2 >> 1;
#pragma unroll
            for (int r = 0; r < 8; ++r) {
                int m = r + 8 * lq;                      // D row of this lane
                float v = acc[ms][i][r];
                v = 0.5f * v * (1.0f + erff(v * 0.70710678118f));
                *(unsigned short*)(Hs + ms * 16384 + kstep * 1024 + pc * 512 +
                                   (m + 16 * hi) * 16 + (jn & 7) * 2) = f2bf(v);
            }
        }
    }
    __syncthreads();

    // ---- FC2: Y[32,2048] += H[32,512] * W2_b^T; 4 groups of 4 N-tiles ----
    for (int g = 0; g < 4; ++g) {
        f32x8 acc2[MSUB][4];
#pragma unroll
        for (int ms = 0; ms < MSUB; ++ms)
#pragma unroll
            for (int i = 0; i < 4; ++i) acc2[ms][i] = (f32x8)0.0f;

        for (int ks = 0; ks < 16; ++ks) {
            AFrag A[MSUB];
#pragma unroll
            for (int ms = 0; ms < MSUB; ++ms) {
                A[ms].q[0] = *(const u32x4*)(Hs + ms * 16384 + ks * 1024 +       lane * 16);
                A[ms].q[1] = *(const u32x4*)(Hs + ms * 16384 + ks * 1024 + 512 + lane * 16);
            }
#pragma unroll
            for (int i = 0; i < 4; ++i) {
                int f = (wv * 16 + g * 4 + i) * 16 + lh;     // W2 row = B column
                const u32x4* bp = (const u32x4*)(W2bf + (size_t)f * OUT_F +
                                                 b * BS + ks * 32 + lq * 8);
                AFrag B; B.q[0] = bp[0]; B.q[1] = bp[2];
#pragma unroll
                for (int ms = 0; ms < MSUB; ++ms)            // B reused x2
                    acc2[ms][i] = __builtin_amdgcn_wmma_f32_16x16x32_bf16(
                        false, A[ms].v, false, B.v, (short)0, acc2[ms][i], false, false);
            }
        }
#pragma unroll
        for (int i = 0; i < 4; ++i) {
            int f = (wv * 16 + g * 4 + i) * 16 + lh;
#pragma unroll
            for (int ms = 0; ms < MSUB; ++ms) {
#pragma unroll
                for (int r = 0; r < 8; ++r) {
                    int m = ms * 16 + r + 8 * lq;
                    if (m < nvalid)
                        atomicAdd(y + (size_t)tok[m] * IN_F + f, acc2[ms][i][r]);
                }
            }
        }
    }
}

// ---------------------------------------------------------------------------
// Workspace layout (bytes):
//   [0,64)            : int cnt[16]
//   [256, 256+512K)   : int list[16][8192]
//   [1M, 1M+32M)      : W1 as bf16  (8192*2048)
//   [1M+32M, 1M+64M)  : W2 as bf16  (2048*8192)   total ~68 MB required
// ---------------------------------------------------------------------------
extern "C" void kernel_launch(void* const* d_in, const int* in_sizes, int n_in,
                              void* d_out, int out_size, void* d_ws, size_t ws_size,
                              hipStream_t stream) {
    (void)in_sizes; (void)n_in; (void)ws_size;
    const float* x  = (const float*)d_in[0];
    const float* Wr = (const float*)d_in[1];
    const float* br = (const float*)d_in[2];
    const float* W1 = (const float*)d_in[3];
    const float* W2 = (const float*)d_in[4];
    float* y = (float*)d_out;

    char* ws = (char*)d_ws;
    int* cnt  = (int*)ws;
    int* list = (int*)(ws + 256);
    unsigned short* W1bf = (unsigned short*)(ws + (1u << 20));
    unsigned short* W2bf = (unsigned short*)(ws + (1u << 20) + (size_t)OUT_F * IN_F * 2);

    hipMemsetAsync(cnt, 0, 64, stream);                              // expert counts
    hipMemsetAsync(y, 0, (size_t)out_size * sizeof(float), stream);  // y accumulated atomically

    const int nW = OUT_F * IN_F;                                     // 16.7M elems each
    convert_bf16<<<nW / (256 * 8), 256, 0, stream>>>(W1, W1bf, nW);
    convert_bf16<<<nW / (256 * 8), 256, 0, stream>>>(W2, W2bf, nW);

    router_topk<<<NTOK / 8, 256, 0, stream>>>(x, Wr, br, cnt, list);

    ffn_kernel<<<dim3(MAX_TILES, NB), 256, SMEM_BYTES, stream>>>(x, W1bf, W2bf,
                                                                 cnt, list, y);
}
